// FraudGraphSAGE_103079215657
// MI455X (gfx1250) — matrix operand order repaired and checked by reading, not compiled, not run on hardware
//
#include <hip/hip_runtime.h>
#include <hip/hip_bf16.h>
#include <stdint.h>

// ---------------------------------------------------------------------------
// Types for CDNA5 WMMA (wave32): V_WMMA_F32_16X16X32_BF16
// ---------------------------------------------------------------------------
typedef __bf16 v16bf __attribute__((ext_vector_type(16)));
typedef float  v8f   __attribute__((ext_vector_type(8)));

union ABFrag {           // 8 VGPRs = 16 bf16 elements per lane
    v16bf v;
    uint32_t u[8];
    uint4 q[2];
};

__device__ __forceinline__ uint32_t f2bf_rne(float f) {
    uint32_t u = __float_as_uint(f);
    return (u + 0x7FFFu + ((u >> 16) & 1u)) >> 16;   // round-to-nearest-even
}
__device__ __forceinline__ uint32_t pack_bf2(float lo, float hi) {
    return f2bf_rne(lo) | (f2bf_rne(hi) << 16);
}

#define CH   128          // channels (hidden == in)
#define ROWS 64           // rows per block for GEMM kernels
#define APITCH 132        // LDS row pitch in dwords (16B aligned, bank-stride 4)

// ---------------------------------------------------------------------------
// Utility kernels
// ---------------------------------------------------------------------------
__global__ void k_zero_f32(float* __restrict__ p, long n) {
    long i = (long)blockIdx.x * blockDim.x + threadIdx.x;
    if (i < n) p[i] = 0.0f;
}

// degree: one atomic add per edge (deg buffer pre-zeroed)
__global__ void k_degree(const int* __restrict__ ei, float* __restrict__ deg,
                         int E) {
    int e = blockIdx.x * blockDim.x + threadIdx.x;
    if (e >= E) return;
    int d = ei[E + e];
    __hip_atomic_fetch_add(deg + d, 1.0f, __ATOMIC_RELAXED, __HIP_MEMORY_SCOPE_AGENT);
}

// deg -> 1/max(deg,1), in place
__global__ void k_rdeg(float* __restrict__ deg, int n) {
    int i = blockIdx.x * blockDim.x + threadIdx.x;
    if (i < n) deg[i] = 1.0f / fmaxf(deg[i], 1.0f);
}

// scatter-sum: 32 threads per edge, 4 channels each (float4 gather, f32 atomics)
__global__ void k_scatter(const float* __restrict__ h, const int* __restrict__ ei,
                          float* __restrict__ agg, int E) {
    long idx = (long)blockIdx.x * blockDim.x + threadIdx.x;
    int e = (int)(idx >> 5);
    int g = (int)(idx & 31);
    if (e >= E) return;
    int s = ei[e];
    int d = ei[E + e];
    const float4 v = *(const float4*)(h + (long)s * CH + g * 4);
    float* a = agg + (long)d * CH + g * 4;
    __hip_atomic_fetch_add(a + 0, v.x, __ATOMIC_RELAXED, __HIP_MEMORY_SCOPE_AGENT);
    __hip_atomic_fetch_add(a + 1, v.y, __ATOMIC_RELAXED, __HIP_MEMORY_SCOPE_AGENT);
    __hip_atomic_fetch_add(a + 2, v.z, __ATOMIC_RELAXED, __HIP_MEMORY_SCOPE_AGENT);
    __hip_atomic_fetch_add(a + 3, v.w, __ATOMIC_RELAXED, __HIP_MEMORY_SCOPE_AGENT);
}

// Pack [wn;wr] (256x128 f32, row-major K x N) into bf16, transposed:
// out[n*128 + p] packs K = 2p, 2p+1 for column n.  p<64 -> wn, p>=64 -> wr.
__global__ void k_prep_w(const float* __restrict__ wn, const float* __restrict__ wr,
                         uint32_t* __restrict__ out) {
    int i = blockIdx.x * blockDim.x + threadIdx.x;     // 128*128 threads
    if (i >= CH * CH) return;
    int n = i >> 7, p = i & 127;
    const float* W = (p < 64) ? wn : wr;
    int k0 = (p < 64) ? 2 * p : 2 * p - CH;
    out[n * CH + p] = pack_bf2(W[k0 * CH + n], W[(k0 + 1) * CH + n]);
}

// cw1 (128x64 f32, K x N) -> bf16 transposed: out[n*64 + p] packs K=2p,2p+1
__global__ void k_prep_c(const float* __restrict__ cw1, uint32_t* __restrict__ out) {
    int i = blockIdx.x * blockDim.x + threadIdx.x;     // 64*64 threads
    if (i >= 64 * 64) return;
    int n = i >> 6, p = i & 63;
    int k0 = 2 * p;
    out[n * 64 + p] = pack_bf2(cw1[k0 * 64 + n], cw1[(k0 + 1) * 64 + n]);
}

// ---------------------------------------------------------------------------
// Fused SAGE layer:  hout = epi( [agg*rdeg | hin] @ Wt  + bb )
//   Wt: bf16 packed, [n(0..127)][kpair(0..127)]  (K = 256 = agg|hin concat)
//   epi: optional BatchNorm(eval) + ReLU
// Block: 256 threads = 8 waves; 64 rows x 128 cols; wave w -> col tile w.
// ---------------------------------------------------------------------------
template <bool BN>
__global__ void __launch_bounds__(256)
k_sage(const float* __restrict__ hin, const float* __restrict__ agg,
       const float* __restrict__ rdeg, const uint32_t* __restrict__ Wt,
       const float* __restrict__ bb,
       const float* __restrict__ g, const float* __restrict__ be,
       const float* __restrict__ rm, const float* __restrict__ rv,
       float* __restrict__ hout, int N) {
    __shared__ uint32_t lds_a[ROWS * APITCH];          // 64 x 256 bf16 (packed pairs)

    const int tid  = threadIdx.x;
    const int lane = tid & 31;
    const int wave = tid >> 5;
    const int half = lane >> 4;                        // 0: lanes 0-15, 1: lanes 16-31
    const int l16  = lane & 15;
    const long rowBase = (long)blockIdx.x * ROWS;

    // ---- stage A tile into LDS as bf16 pairs: cols[0..63]=agg*rdeg, [64..127]=hin
    for (int i = tid; i < ROWS * 128; i += 256) {
        int r = i >> 7, p = i & 127;
        long row = rowBase + r;
        uint32_t val = 0;
        if (row < N) {
            float2 f;
            if (p < 64) {
                f = *(const float2*)(agg + row * CH + 2 * p);
                float s = rdeg[row];
                f.x *= s; f.y *= s;
            } else {
                f = *(const float2*)(hin + row * CH + 2 * (p - 64));
            }
            val = pack_bf2(f.x, f.y);
        }
        lds_a[r * APITCH + p] = val;
    }
    __syncthreads();

    // ---- WMMA: 4 row tiles x 1 col tile per wave, K = 256 in 8 steps of 32
    const int n0 = wave * 16;
    v8f acc[4] = {};
    for (int kk = 0; kk < 8; ++kk) {
        ABFrag b;
        const uint4* bp = (const uint4*)(Wt + (n0 + l16) * CH + kk * 16 + half * 4);
        b.q[0] = bp[0];
        b.q[1] = bp[2];
        for (int rt = 0; rt < 4; ++rt) {
            ABFrag a;
            const uint32_t* rp = lds_a + (rt * 16 + l16) * APITCH + kk * 16 + half * 4;
            a.q[0] = *(const uint4*)(rp);
            a.q[1] = *(const uint4*)(rp + 8);
            acc[rt] = __builtin_amdgcn_wmma_f32_16x16x32_bf16(
                false, a.v, false, b.v, (short)0, acc[rt], false, false);
        }
    }

    // ---- epilogue: +bias, optional BN(eval)+ReLU, store f32
    const int n = n0 + l16;
    const float bias = bb[n];
    float scale = 1.0f, shift = 0.0f;
    if (BN) {
        float s = g[n] * rsqrtf(rv[n] + 1e-5f);
        scale = s;
        shift = be[n] - rm[n] * s;
    }
    for (int rt = 0; rt < 4; ++rt) {
        for (int r = 0; r < 8; ++r) {
            long m = rowBase + rt * 16 + r + half * 8;
            if (m < N) {
                float v = acc[rt][r] + bias;
                if (BN) v = fmaxf(v * scale + shift, 0.0f);
                hout[m * CH + n] = v;
            }
        }
    }
}

// ---------------------------------------------------------------------------
// Classifier: out = relu(h @ cw1 + cb1) @ cw2 + cb2
//   ct1: bf16 packed [n(0..63)][kpair(0..63)]
// Block: 256 threads = 8 waves, 64 rows. Wave w -> col tile w&3, row half w>>2.
// ---------------------------------------------------------------------------
#define CPITCH 68
__global__ void __launch_bounds__(256)
k_classifier(const float* __restrict__ hin, const uint32_t* __restrict__ ct1,
             const float* __restrict__ cb1, const float* __restrict__ cw2,
             const float* __restrict__ cb2, float* __restrict__ out, int N) {
    __shared__ uint32_t lds_a[ROWS * CPITCH];   // 64 x 128 bf16
    __shared__ float    zbuf[ROWS * CPITCH];    // 64 x 64 f32 (+pad)

    const int tid  = threadIdx.x;
    const int lane = tid & 31;
    const int wave = tid >> 5;
    const int half = lane >> 4;
    const int l16  = lane & 15;
    const long rowBase = (long)blockIdx.x * ROWS;

    // stage h rows as bf16 pairs
    for (int i = tid; i < ROWS * 64; i += 256) {
        int r = i >> 6, p = i & 63;
        long row = rowBase + r;
        uint32_t val = 0;
        if (row < N) {
            float2 f = *(const float2*)(hin + row * CH + 2 * p);
            val = pack_bf2(f.x, f.y);
        }
        lds_a[r * CPITCH + p] = val;
    }
    __syncthreads();

    // GEMM1: z = relu(h @ cw1 + cb1), K = 128 in 4 steps
    const int ct = wave & 3;
    const int rh = wave >> 2;
    const int n0 = ct * 16;
    v8f acc[2] = {};
    for (int kk = 0; kk < 4; ++kk) {
        ABFrag b;
        const uint4* bp = (const uint4*)(ct1 + (n0 + l16) * 64 + kk * 16 + half * 4);
        b.q[0] = bp[0];
        b.q[1] = bp[2];
        for (int j = 0; j < 2; ++j) {
            int rt = rh * 2 + j;
            ABFrag a;
            const uint32_t* rp = lds_a + (rt * 16 + l16) * CPITCH + kk * 16 + half * 4;
            a.q[0] = *(const uint4*)(rp);
            a.q[1] = *(const uint4*)(rp + 8);
            acc[j] = __builtin_amdgcn_wmma_f32_16x16x32_bf16(
                false, a.v, false, b.v, (short)0, acc[j], false, false);
        }
    }
    const int n = n0 + l16;
    const float cb = cb1[n];
    for (int j = 0; j < 2; ++j) {
        int rt = rh * 2 + j;
        for (int r = 0; r < 8; ++r) {
            int m = rt * 16 + r + half * 8;
            zbuf[m * CPITCH + n] = fmaxf(acc[j][r] + cb, 0.0f);
        }
    }
    __syncthreads();

    // GEMM2: tiny 64->2 projection, one thread per (row, out-ch)
    if (tid < ROWS * 2) {
        int m = tid >> 1, o = tid & 1;
        float sum = cb2[o];
        const float* zr = zbuf + m * CPITCH;
        for (int k = 0; k < 64; ++k) sum += zr[k] * cw2[k * 2 + o];
        long gm = rowBase + m;
        if (gm < N) out[gm * 2 + o] = sum;
    }
}

// ---------------------------------------------------------------------------
// Launch
// ---------------------------------------------------------------------------
static inline size_t alignup(size_t x) { return (x + 255) & ~(size_t)255; }

extern "C" void kernel_launch(void* const* d_in, const int* in_sizes, int n_in,
                              void* d_out, int out_size, void* d_ws, size_t ws_size,
                              hipStream_t stream) {
    const float* x   = (const float*)d_in[0];
    const int*   ei  = (const int*)d_in[1];
    const float* wn0 = (const float*)d_in[2];
    const float* bb0 = (const float*)d_in[3];
    const float* wr0 = (const float*)d_in[4];
    const float* wn1 = (const float*)d_in[5];
    const float* bb1 = (const float*)d_in[6];
    const float* wr1 = (const float*)d_in[7];
    const float* wn2 = (const float*)d_in[8];
    const float* bb2 = (const float*)d_in[9];
    const float* wr2 = (const float*)d_in[10];
    const float* g0  = (const float*)d_in[11];
    const float* be0 = (const float*)d_in[12];
    const float* rm0 = (const float*)d_in[13];
    const float* rv0 = (const float*)d_in[14];
    const float* g1  = (const float*)d_in[15];
    const float* be1 = (const float*)d_in[16];
    const float* rm1 = (const float*)d_in[17];
    const float* rv1 = (const float*)d_in[18];
    const float* cw1 = (const float*)d_in[19];
    const float* cb1 = (const float*)d_in[20];
    const float* cw2 = (const float*)d_in[21];
    const float* cb2 = (const float*)d_in[22];
    float* out = (float*)d_out;

    const int N = in_sizes[0] / CH;
    const int E = in_sizes[1] / 2;

    // workspace layout
    char* ws = (char*)d_ws;
    size_t off = 0;
    float* agg  = (float*)(ws + off); off = alignup(off + (size_t)N * CH * 4);
    float* hA   = (float*)(ws + off); off = alignup(off + (size_t)N * CH * 4);
    float* hB   = (float*)(ws + off); off = alignup(off + (size_t)N * CH * 4);
    float* rdeg = (float*)(ws + off); off = alignup(off + (size_t)N * 4);
    uint32_t* Wt0 = (uint32_t*)(ws + off); off = alignup(off + (size_t)CH * CH * 4);
    uint32_t* Wt1 = (uint32_t*)(ws + off); off = alignup(off + (size_t)CH * CH * 4);
    uint32_t* Wt2 = (uint32_t*)(ws + off); off = alignup(off + (size_t)CH * CH * 4);
    uint32_t* ct1 = (uint32_t*)(ws + off); off = alignup(off + (size_t)64 * 64 * 4);

    const int gemmBlocks = (N + ROWS - 1) / ROWS;
    const long featElems = (long)N * CH;
    const int zeroBlocksFeat = (int)((featElems + 255) / 256);
    const int scatBlocks = (int)(((long)E * 32 + 255) / 256);

    // weight repack (bf16, transposed)
    k_prep_w<<<(CH * CH + 255) / 256, 256, 0, stream>>>(wn0, wr0, Wt0);
    k_prep_w<<<(CH * CH + 255) / 256, 256, 0, stream>>>(wn1, wr1, Wt1);
    k_prep_w<<<(CH * CH + 255) / 256, 256, 0, stream>>>(wn2, wr2, Wt2);
    k_prep_c<<<(64 * 64 + 255) / 256, 256, 0, stream>>>(cw1, ct1);

    // degrees -> reciprocal
    k_zero_f32<<<(N + 255) / 256, 256, 0, stream>>>(rdeg, N);
    k_degree<<<(E + 255) / 256, 256, 0, stream>>>(ei, rdeg, E);
    k_rdeg<<<(N + 255) / 256, 256, 0, stream>>>(rdeg, N);

    // layer 0: x -> hA   (BN0 + ReLU)
    k_zero_f32<<<zeroBlocksFeat, 256, 0, stream>>>(agg, featElems);
    k_scatter<<<scatBlocks, 256, 0, stream>>>(x, ei, agg, E);
    k_sage<true><<<gemmBlocks, 256, 0, stream>>>(x, agg, rdeg, Wt0, bb0,
                                                 g0, be0, rm0, rv0, hA, N);
    // layer 1: hA -> hB  (BN1 + ReLU)
    k_zero_f32<<<zeroBlocksFeat, 256, 0, stream>>>(agg, featElems);
    k_scatter<<<scatBlocks, 256, 0, stream>>>(hA, ei, agg, E);
    k_sage<true><<<gemmBlocks, 256, 0, stream>>>(hA, agg, rdeg, Wt1, bb1,
                                                 g1, be1, rm1, rv1, hB, N);
    // layer 2: hB -> hA  (no BN / ReLU)
    k_zero_f32<<<zeroBlocksFeat, 256, 0, stream>>>(agg, featElems);
    k_scatter<<<scatBlocks, 256, 0, stream>>>(hB, ei, agg, E);
    k_sage<false><<<gemmBlocks, 256, 0, stream>>>(hB, agg, rdeg, Wt2, bb2,
                                                  bb2, bb2, bb2, bb2, hA, N);
    // classifier
    k_classifier<<<gemmBlocks, 256, 0, stream>>>(hA, ct1, cb1, cw2, cb2, out, N);
}
